// LinearQKVAttention_26371099198131
// MI455X (gfx1250) — matrix operand order repaired and checked by reading, not compiled, not run on hardware
//
#include <hip/hip_runtime.h>
#include <hip/hip_bf16.h>
#include <stdint.h>

// ---------------------------------------------------------------------------
// LinearQKVAttention for MI455X (gfx1250, wave32, WMMA bf16 16x16x32)
//   q,k,v : f32 (8,16,4096,64)   out : f32 (8,16,4096,64)
//
//   res = (softmax_row(q)/8) @ [ (1/s_d) * sum_n exp(k[n,d]-m_d) * v[n,e] ]
// ---------------------------------------------------------------------------

#define BH   128      // batch*heads
#define SEQ  4096
#define DIM  64
#define NCHUNK 8      // context kernel: chunks per head (512 rows each)

typedef __attribute__((ext_vector_type(16))) __bf16 v16bf;
typedef __attribute__((ext_vector_type(8)))  float  v8f;

union BF16x16 { v16bf v; __bf16 e[16]; };

// float -> bf16, round-to-nearest-even, via explicit bit math
static __device__ inline __bf16 f2bf(float f) {
    union { float f; uint32_t u; } in; in.f = f;
    uint32_t u = in.u + 0x7FFFu + ((in.u >> 16) & 1u);
    union { uint16_t s; __bf16 b; } out; out.s = (uint16_t)(u >> 16);
    return out.b;
}

// A-fragment K index for 16-bit 16x32 A matrix (lane half = lane>>4)
static __device__ inline int kmapA(int i, int half) {
    return i + 8 * half + ((i >= 8) ? 8 : 0);
}

// ---------------------------------------------------------------------------
// Kernel 1: per-column (over seq) max and exp-sum of k.  128 blocks x 256.
// Thread t: column d = t&63, sequence slice s = t>>6 (1024 rows each).
// Second pass re-reads k out of L2 (1 MB/head * 128 heads < 192 MB L2).
// ---------------------------------------------------------------------------
__global__ __launch_bounds__(256) void lqkv_colstats(
    const float* __restrict__ K, float* __restrict__ colmax, float* __restrict__ colsum)
{
    const int bh = blockIdx.x;
    const int t  = threadIdx.x;
    const int d  = t & 63;
    const int s  = t >> 6;
    const size_t base = (size_t)bh * SEQ * DIM;
    const float* kp = K + base + (size_t)(s * 1024) * DIM + d;

    __shared__ float red[256];
    __shared__ float gmax[64];

    float m = -3.402823466e38f;
    for (int n = 0; n < 1024; ++n) m = fmaxf(m, kp[n * DIM]);
    red[t] = m;
    __syncthreads();
    if (s == 0) {
        gmax[d] = fmaxf(fmaxf(red[d], red[64 + d]), fmaxf(red[128 + d], red[192 + d]));
    }
    __syncthreads();
    const float M = gmax[d];

    float sum = 0.0f;
    for (int n = 0; n < 1024; ++n) sum += __expf(kp[n * DIM] - M);
    __syncthreads();
    red[t] = sum;
    __syncthreads();
    if (s == 0) {
        colmax[bh * 64 + d] = M;
        colsum[bh * 64 + d] = red[d] + red[64 + d] + red[128 + d] + red[192 + d];
    }
}

// ---------------------------------------------------------------------------
// Kernel 2: partial C[d,e] += (1/s_d) * sum_n exp(k[n,d]-m_d) * v[n,e]
// Grid = BH*NCHUNK blocks, 512 threads (16 waves). Each wave owns one 16x16
// tile of the 64x64 C matrix. 64 seq rows staged per iteration into
// double-buffered LDS (one barrier per iteration), 2 WMMA k-steps per wave.
// ---------------------------------------------------------------------------
__global__ __launch_bounds__(512) void lqkv_context(
    const float* __restrict__ K, const float* __restrict__ V,
    const float* __restrict__ colmax, const float* __restrict__ colsum,
    float* __restrict__ C)
{
    const int bx    = blockIdx.x;
    const int bh    = bx >> 3;
    const int chunk = bx & (NCHUNK - 1);
    const int t     = threadIdx.x;
    const size_t base = (size_t)bh * SEQ * DIM;

    __shared__ __bf16 Et[2][64 * 64];   // Et[p][d][j] = exp(k[n0+j][d]-m_d)
    __shared__ __bf16 Vt[2][64 * 64];   // Vt[p][e][j] = v[n0+j][e]
    __shared__ float  cm[64];
    __shared__ float  rs[64];

    if (t < 64) {
        cm[t] = colmax[bh * 64 + t];
        rs[t] = 1.0f / colsum[bh * 64 + t];
    }

    const int wave = t >> 5, lane = t & 31;
    const int d0 = (wave >> 2) * 16, e0 = (wave & 3) * 16;
    const int half = lane >> 4, lr = lane & 15;
    const int nl = t >> 4;              // staged row 0..31 (+32 second half)
    const int dq = (t & 15) * 4;        // staged col group

    v8f acc = {};
    __syncthreads();

    const int nBeg = chunk * (SEQ / NCHUNK);
    const int nEnd = nBeg + (SEQ / NCHUNK);
    int p = 0;
    for (int n0 = nBeg; n0 < nEnd; n0 += 64) {
        // ---- stage 64x64 tiles of exp(k) and v as bf16 (transposed) ----
        __bf16* et = &Et[p][0];
        __bf16* vt = &Vt[p][0];
        #pragma unroll
        for (int h2 = 0; h2 < 2; ++h2) {
            const int j = nl + 32 * h2;
            const float4 k4 = *(const float4*)(K + base + (size_t)(n0 + j) * DIM + dq);
            const float4 v4 = *(const float4*)(V + base + (size_t)(n0 + j) * DIM + dq);
            et[(dq + 0) * 64 + j] = f2bf(__expf(k4.x - cm[dq + 0]));
            et[(dq + 1) * 64 + j] = f2bf(__expf(k4.y - cm[dq + 1]));
            et[(dq + 2) * 64 + j] = f2bf(__expf(k4.z - cm[dq + 2]));
            et[(dq + 3) * 64 + j] = f2bf(__expf(k4.w - cm[dq + 3]));
            vt[(dq + 0) * 64 + j] = f2bf(v4.x);
            vt[(dq + 1) * 64 + j] = f2bf(v4.y);
            vt[(dq + 2) * 64 + j] = f2bf(v4.z);
            vt[(dq + 3) * 64 + j] = f2bf(v4.w);
        }
        // speculative prefetch of next tile (global_prefetch_b8; safely dropped OOB)
        __builtin_prefetch(K + base + (size_t)(n0 + 64 + nl) * DIM + dq, 0, 1);
        __builtin_prefetch(V + base + (size_t)(n0 + 64 + nl) * DIM + dq, 0, 1);
        __syncthreads();

        // ---- WMMA: A = Expᵀ tile (16 d x 32 n), B = V tile (32 n x 16 e) ----
        #pragma unroll
        for (int ks = 0; ks < 2; ++ks) {
            const int kb = ks * 32;
            BF16x16 a, b;
            #pragma unroll
            for (int i = 0; i < 16; ++i) a.e[i] = et[(d0 + lr) * 64 + kb + kmapA(i, half)];
            #pragma unroll
            for (int i = 0; i < 16; ++i) b.e[i] = vt[(e0 + lr) * 64 + kb + kmapA(i, half)];
            acc = __builtin_amdgcn_wmma_f32_16x16x32_bf16(
                false, a.v, false, b.v, (short)0, acc, false, false);
        }
        p ^= 1;   // double buffer: buffer p re-staged only after the NEXT barrier
    }

    // ---- scale rows by 1/colsum and accumulate partial C ----
    #pragma unroll
    for (int r = 0; r < 8; ++r) {
        const int M = r + 8 * half;                    // D-matrix row layout
        const float val = acc[r] * rs[d0 + M];
        unsafeAtomicAdd(&C[(size_t)bh * 4096 + (d0 + M) * 64 + e0 + lr], val);
    }
}

// ---------------------------------------------------------------------------
// Kernel 3: res = rowsoftmax(q)*0.125 @ C.  Grid = BH*(SEQ/64) blocks,
// 256 threads (8 waves). 64-row q tile per block; each wave computes two
// 16x16 output tiles (2 WMMA k-steps each), results staged through LDS for
// fully coalesced b128 global stores.
// ---------------------------------------------------------------------------
__global__ __launch_bounds__(256) void lqkv_qcontext(
    const float* __restrict__ Q, const float* __restrict__ C, float* __restrict__ out)
{
    const int bx = blockIdx.x;
    const int bh = bx >> 6;
    const int nb = (bx & 63) * 64;
    const int t  = threadIdx.x;
    const size_t base = (size_t)bh * SEQ * DIM;

    __shared__ __bf16 Qs[64 * 64];   // Qs[row][d]  (row-major: contiguous in d)
    __shared__ __bf16 Ct[64 * 64];   // Ct[e][d]    (transposed: contiguous in d)
    __shared__ float  outS[64 * 64]; // f32 result staging for coalesced stores
    __shared__ float  redM[256];
    __shared__ float  redS[256];

    // ---- stage C (64x64 f32 -> bf16, transposed): 16 elems per thread ----
    {
        const float* Cp = C + (size_t)bh * 4096 + t * 16;
        #pragma unroll
        for (int i = 0; i < 16; ++i) {
            const int idx = t * 16 + i;
            Ct[(idx & 63) * 64 + (idx >> 6)] = f2bf(Cp[i]);
        }
    }

    // ---- q row softmax: 4 threads per row, 16 channels each ----
    const int r = t >> 2, quarter = t & 3;
    const float* qr = Q + base + (size_t)(nb + r) * DIM + quarter * 16;
    float xs[16];
    #pragma unroll
    for (int i = 0; i < 4; ++i) {
        const float4 f = *(const float4*)(qr + i * 4);
        xs[i * 4 + 0] = f.x; xs[i * 4 + 1] = f.y; xs[i * 4 + 2] = f.z; xs[i * 4 + 3] = f.w;
    }
    float m = -3.402823466e38f;
    #pragma unroll
    for (int i = 0; i < 16; ++i) m = fmaxf(m, xs[i]);
    redM[r * 4 + quarter] = m;
    __syncthreads();
    const float rm = fmaxf(fmaxf(redM[r * 4 + 0], redM[r * 4 + 1]),
                           fmaxf(redM[r * 4 + 2], redM[r * 4 + 3]));
    float psum = 0.0f;
    #pragma unroll
    for (int i = 0; i < 16; ++i) psum += __expf(xs[i] - rm);
    redS[r * 4 + quarter] = psum;
    __syncthreads();
    const float rsum = redS[r * 4 + 0] + redS[r * 4 + 1] + redS[r * 4 + 2] + redS[r * 4 + 3];
    const float sc = 0.125f / rsum;   // fold in 1/sqrt(64)
    #pragma unroll
    for (int i = 0; i < 16; ++i)
        Qs[r * 64 + quarter * 16 + i] = f2bf(__expf(xs[i] - rm) * sc);
    __syncthreads();

    // ---- WMMA: each wave -> 2 output tiles, 2 k-steps each ----
    const int wave = t >> 5, lane = t & 31;
    const int half = lane >> 4, lr = lane & 15;
    #pragma unroll
    for (int tt = 0; tt < 2; ++tt) {
        const int tile = wave + tt * 8;
        const int m0 = (tile >> 2) * 16, e0 = (tile & 3) * 16;
        v8f acc = {};
        #pragma unroll
        for (int ks = 0; ks < 2; ++ks) {
            const int kb = ks * 32;
            BF16x16 a, b;
            #pragma unroll
            for (int i = 0; i < 16; ++i) a.e[i] = Qs[(m0 + lr) * 64 + kb + kmapA(i, half)];
            #pragma unroll
            for (int i = 0; i < 16; ++i) b.e[i] = Ct[(e0 + lr) * 64 + kb + kmapA(i, half)];
            acc = __builtin_amdgcn_wmma_f32_16x16x32_bf16(
                false, a.v, false, b.v, (short)0, acc, false, false);
        }
        #pragma unroll
        for (int rr = 0; rr < 8; ++rr) {
            const int M = rr + 8 * half;
            outS[(m0 + M) * 64 + e0 + lr] = acc[rr];
        }
    }
    __syncthreads();

    // ---- coalesced b128 stores: thread t writes a quarter of row t>>2 ----
    {
        const int row = t >> 2, col = (t & 3) * 16;
        float* op = out + base + (size_t)(nb + row) * DIM + col;
        #pragma unroll
        for (int i = 0; i < 4; ++i)
            *(float4*)(op + i * 4) = *(const float4*)&outS[row * 64 + col + i * 4];
    }
}

// ---------------------------------------------------------------------------
extern "C" void kernel_launch(void* const* d_in, const int* in_sizes, int n_in,
                              void* d_out, int out_size, void* d_ws, size_t ws_size,
                              hipStream_t stream) {
    const float* q = (const float*)d_in[0];
    const float* k = (const float*)d_in[1];
    const float* v = (const float*)d_in[2];
    float* out = (float*)d_out;

    float* wsf    = (float*)d_ws;
    float* colmax = wsf;                 // BH*64
    float* colsum = wsf + BH * 64;       // BH*64
    float* Cmat   = wsf + 2 * BH * 64;   // BH*64*64

    hipMemsetAsync(Cmat, 0, (size_t)BH * 4096 * sizeof(float), stream);

    lqkv_colstats<<<BH, 256, 0, stream>>>(k, colmax, colsum);
    lqkv_context<<<BH * NCHUNK, 512, 0, stream>>>(k, v, colmax, colsum, Cmat);
    lqkv_qcontext<<<BH * (SEQ / 64), 256, 0, stream>>>(q, Cmat, out);
}
